// GAT_40114994545116
// MI455X (gfx1250) — matrix operand order repaired and checked by reading, not compiled, not run on hardware
//
#include <hip/hip_runtime.h>
#include <hip/hip_bf16.h>

// ---------------------------------------------------------------------------
// GAT on MI455X (gfx1250): bf16 WMMA GEMMs (2x2 register tile per wave,
// vectorized b128 staging) + L2-resident edge softmax/scatter
// ---------------------------------------------------------------------------

#define GN 50000
#define GE 800000
#define GIN 128
#define GT 8
#define GH 4
#define GF 64
#define GC 256
#define GEND 256

typedef __attribute__((ext_vector_type(16))) __bf16          v16bf;
typedef __attribute__((ext_vector_type(16))) unsigned short  v16u;
typedef __attribute__((ext_vector_type(8)))  float           v8f;

struct U32x8 { uint4 a, b; };   // 32 bytes, bit-castable to v16bf

__device__ __forceinline__ unsigned short f2bf(float f) {
    unsigned u = __float_as_uint(f);
    return (unsigned short)((u + 0x7FFFu + ((u >> 16) & 1u)) >> 16); // RNE
}
__device__ __forceinline__ unsigned pack2bf(float lo, float hi) {
    return (unsigned)f2bf(lo) | ((unsigned)f2bf(hi) << 16);
}

// monotonic float <-> uint for atomic segment-max
__device__ __forceinline__ unsigned f2ord(float f) {
    unsigned u = __float_as_uint(f);
    return (u & 0x80000000u) ? ~u : (u | 0x80000000u);
}
__device__ __forceinline__ float ord2f(unsigned u) {
    unsigned v = (u & 0x80000000u) ? (u & 0x7FFFFFFFu) : ~u;
    return __uint_as_float(v);
}

// ---------------------------------------------------------------------------
// Pack fp32 weight [K x Nout] into WMMA-B-fragment-major bf16:
//   P[(tk*nTiles + tn)*512 + lane*16 + e] = bf16( W[k][n] )
//   k = tk*32 + (lane>=16 ? 16 : 0) + e ;  n = tn*16 + (lane&15)
// ---------------------------------------------------------------------------
__global__ void pack_w_kernel(const float* __restrict__ W, int K, int Nout,
                              unsigned short* __restrict__ P) {
    int t = blockIdx.x * blockDim.x + threadIdx.x;
    if (t >= K * Nout) return;
    int e    = t & 15;
    int lane = (t >> 4) & 31;
    int nt   = Nout >> 4;
    int tn   = (t >> 9) % nt;
    int tk   = t / (nt << 9);
    int k = (tk << 5) + ((lane >> 4) << 4) + e;
    int n = (tn << 4) + (lane & 15);
    P[t] = f2bf(W[k * Nout + n]);
}

// ---------------------------------------------------------------------------
// GEMM: out[M x Nout] = A[M x K] (fp32, generic strides) @ Wp (packed bf16)
// + optional bias / relu.
// Block: 256 threads (8 waves) -> 128 rows x 64 cols tile.
// Wave (rowGrp = wave>>1, colGrp = wave&1) -> 32 rows x 32 cols:
// 2 A-frags x 2 B-frags -> 4 WMMAs per 32-wide K step.
// A access: A[row*aRowStride + k*aKStride + aKOff]  (fuses X[:,:,-1] slice).
// contig==1 (aKStride==1, aKOff==0): vectorized float4 / ds_store_b128 path.
// ---------------------------------------------------------------------------
__global__ __launch_bounds__(256)
void gemm_bf16_wmma(const float* __restrict__ A, int M, int K, int Nout,
                    int aRowStride, int aKStride, int aKOff,
                    const unsigned short* __restrict__ Wp,
                    const float* __restrict__ bias,
                    float* __restrict__ out, int relu, int contig) {
    __shared__ __align__(16) unsigned short As[128 * 32];   // 8 KB bf16 tile

    const int tid    = threadIdx.x;
    const int wave   = tid >> 5;
    const int lane   = tid & 31;
    const int hiHalf = lane >> 4;                 // 0 or 1
    const int rowBase = blockIdx.y * 128;
    const int nTiles  = Nout >> 4;

    v8f acc00 = {}, acc01 = {}, acc10 = {}, acc11 = {};

    const int tnBase = blockIdx.x * 4 + (wave & 1) * 2;     // 16-col tile units
    const int rowl0  = (wave >> 1) * 32 + (lane & 15);      // A-frag rows
    const int rowl1  = rowl0 + 16;
    const int aBase  = hiHalf ? 16 : 0;           // K sub-offset (contiguous k = 16h + s)

    // vectorized-staging thread mapping: row = tid/2, 16-col half = tid&1
    const int vRow = tid >> 1;
    const int vKK  = (tid & 1) << 4;
    int vGr = rowBase + vRow; if (vGr >= M) vGr = M - 1;

    for (int kc = 0; kc < K; kc += 32) {
        if (contig) {
            // 16 contiguous floats per thread: 4x global_load_b128,
            // packed bf16 out as 2x ds_store_b128
            const float* sp = A + (size_t)vGr * aRowStride + kc + vKK;
            float4 v0 = *(const float4*)(sp + 0);
            float4 v1 = *(const float4*)(sp + 4);
            float4 v2 = *(const float4*)(sp + 8);
            float4 v3 = *(const float4*)(sp + 12);
            uint4 w0, w1;
            w0.x = pack2bf(v0.x, v0.y); w0.y = pack2bf(v0.z, v0.w);
            w0.z = pack2bf(v1.x, v1.y); w0.w = pack2bf(v1.z, v1.w);
            w1.x = pack2bf(v2.x, v2.y); w1.y = pack2bf(v2.z, v2.w);
            w1.z = pack2bf(v3.x, v3.y); w1.w = pack2bf(v3.z, v3.w);
            *(uint4*)&As[vRow * 32 + vKK]     = w0;
            *(uint4*)&As[vRow * 32 + vKK + 8] = w1;
        } else {
            // generic strided path (X[:,:,-1] slice)
            #pragma unroll
            for (int i = 0; i < 16; ++i) {
                int idx = i * 256 + tid;          // = r*32 + kk
                int r  = idx >> 5;
                int kk = idx & 31;
                int gr = rowBase + r; if (gr >= M) gr = M - 1;
                float v = A[gr * aRowStride + (kc + kk) * aKStride + aKOff];
                As[idx] = f2bf(v);
            }
        }
        // prefetch next K-slice of A (one line per row when contiguous)
        if (kc + 32 < K && tid < 128) {
            int gr = rowBase + tid; if (gr >= M) gr = M - 1;
            __builtin_prefetch(&A[gr * aRowStride + (kc + 32) * aKStride + aKOff], 0, 0);
        }
        __syncthreads();

        // A fragments (contiguous k order matching packed B: lanes<16 -> K{0-15},
        // lanes>=16 -> K{16-31}; two contiguous 16B LDS loads each)
        const unsigned short* ap0 = &As[rowl0 * 32 + aBase];
        const unsigned short* ap1 = &As[rowl1 * 32 + aBase];
        U32x8 at0, at1;
        at0.a = *(const uint4*)(ap0);
        at0.b = *(const uint4*)(ap0 + 8);
        at1.a = *(const uint4*)(ap1);
        at1.b = *(const uint4*)(ap1 + 8);
        v16bf a0 = __builtin_bit_cast(v16bf, at0);
        v16bf a1 = __builtin_bit_cast(v16bf, at1);

        // B fragments: contiguous 32B loads from packed weights
        const unsigned short* bp0 =
            Wp + (((kc >> 5) * nTiles + tnBase) << 9) + (lane << 4);
        v16u bu0 = *(const v16u*)bp0;
        v16u bu1 = *(const v16u*)(bp0 + 512);
        v16bf b0 = __builtin_bit_cast(v16bf, bu0);
        v16bf b1 = __builtin_bit_cast(v16bf, bu1);

        acc00 = __builtin_amdgcn_wmma_f32_16x16x32_bf16(false, a0, false, b0,
                                                        (short)0, acc00, false, false);
        acc01 = __builtin_amdgcn_wmma_f32_16x16x32_bf16(false, a0, false, b1,
                                                        (short)0, acc01, false, false);
        acc10 = __builtin_amdgcn_wmma_f32_16x16x32_bf16(false, a1, false, b0,
                                                        (short)0, acc10, false, false);
        acc11 = __builtin_amdgcn_wmma_f32_16x16x32_bf16(false, a1, false, b1,
                                                        (short)0, acc11, false, false);
        __syncthreads();
    }

    // C/D layout: lanes<16 N=lane, M=r; lanes>=16 N=lane-16, M=8+r
    const int col0    = (tnBase << 4) + (lane & 15);
    const int col1    = col0 + 16;
    const int rowTop0 = rowBase + (wave >> 1) * 32 + (hiHalf << 3);
    const int rowTop1 = rowTop0 + 16;
    const float bv0   = bias ? bias[col0] : 0.0f;
    const float bv1   = bias ? bias[col1] : 0.0f;

    #pragma unroll
    for (int r = 0; r < 8; ++r) {
        int row = rowTop0 + r;
        if (row < M) {
            float v = acc00[r] + bv0;
            if (relu && v < 0.0f) v = 0.0f;
            out[row * Nout + col0] = v;
            float w = acc01[r] + bv1;
            if (relu && w < 0.0f) w = 0.0f;
            out[row * Nout + col1] = w;
        }
    }
    #pragma unroll
    for (int r = 0; r < 8; ++r) {
        int row = rowTop1 + r;
        if (row < M) {
            float v = acc10[r] + bv0;
            if (relu && v < 0.0f) v = 0.0f;
            out[row * Nout + col0] = v;
            float w = acc11[r] + bv1;
            if (relu && w < 0.0f) w = 0.0f;
            out[row * Nout + col1] = w;
        }
    }
}

// ---------------------------------------------------------------------------
// Attention coefficients: a_s[n,h] = <h[n,h,:], att_s[h,:]>, same for a_d
// ---------------------------------------------------------------------------
__global__ void att_kernel(const float* __restrict__ h,
                           const float* __restrict__ atts,
                           const float* __restrict__ attd,
                           float* __restrict__ aS, float* __restrict__ aD,
                           int n) {
    int t = blockIdx.x * blockDim.x + threadIdx.x;
    if (t >= n * GH) return;
    int node = t >> 2, hh = t & 3;
    const float* hp = h + node * GC + hh * GF;
    const float* sp = atts + hh * GF;
    const float* dp = attd + hh * GF;
    float ss = 0.f, dd = 0.f;
    #pragma unroll 8
    for (int f = 0; f < GF; ++f) { float v = hp[f]; ss += v * sp[f]; dd += v * dp[f]; }
    aS[t] = ss; aD[t] = dd;
}

// ---------------------------------------------------------------------------
// Edge pass 1: per-destination segment max (atomic on ordered-uint encoding)
// ---------------------------------------------------------------------------
__global__ void edge_max_kernel(const int* __restrict__ src, const int* __restrict__ dst,
                                const float* __restrict__ aS, const float* __restrict__ aD,
                                unsigned* __restrict__ menc, int e) {
    int t = blockIdx.x * blockDim.x + threadIdx.x;
    if (t >= e * GH) return;
    int ed = t >> 2, hh = t & 3;
    int s = src[ed], d = dst[ed];
    float v = aS[s * GH + hh] + aD[d * GH + hh];
    v = v > 0.f ? v : 0.2f * v;                    // leaky_relu(0.2)
    atomicMax(&menc[d * GH + hh], f2ord(v));
}

// ---------------------------------------------------------------------------
// Edge pass 2: ex = exp(e - m[dst]); segment sum
// ---------------------------------------------------------------------------
__global__ void edge_exp_kernel(const int* __restrict__ src, const int* __restrict__ dst,
                                const float* __restrict__ aS, const float* __restrict__ aD,
                                const unsigned* __restrict__ menc,
                                float* __restrict__ ssum, float* __restrict__ ex, int e) {
    int t = blockIdx.x * blockDim.x + threadIdx.x;
    if (t >= e * GH) return;
    int ed = t >> 2, hh = t & 3;
    int s = src[ed], d = dst[ed];
    float v = aS[s * GH + hh] + aD[d * GH + hh];
    v = v > 0.f ? v : 0.2f * v;
    unsigned me = menc[d * GH + hh];
    float m = me ? ord2f(me) : 0.0f;               // sentinel: no edges -> 0
    float exv = __expf(v - m);
    ex[t] = exv;
    atomicAdd(&ssum[d * GH + hh], exv);
}

// ---------------------------------------------------------------------------
// Edge pass 3: agg[dst] += h[src] * alpha   (h, agg live in 192MB L2)
// one thread per (edge, head, feature)
// ---------------------------------------------------------------------------
__global__ void edge_agg_kernel(const int* __restrict__ src, const int* __restrict__ dst,
                                const float* __restrict__ h,
                                const float* __restrict__ ex,
                                const float* __restrict__ ssum,
                                float* __restrict__ agg, int e) {
    int t = blockIdx.x * blockDim.x + threadIdx.x;
    if (t >= e * GC) return;
    int ed  = t >> 8;
    int rem = t & 255;
    int hh  = rem >> 6;
    int f   = rem & 63;
    int s = src[ed], d = dst[ed];
    float alpha = ex[ed * GH + hh] / (ssum[d * GH + hh] + 1e-16f);
    atomicAdd(&agg[d * GC + hh * GF + f], h[s * GC + hh * GF + f] * alpha);
}

// ---------------------------------------------------------------------------
// BatchNorm: per-channel mean / rsqrt(var+eps)  (one block per channel)
// ---------------------------------------------------------------------------
__global__ void bn_reduce_kernel(const float* __restrict__ x, int n, int c,
                                 float* __restrict__ mu, float* __restrict__ rs) {
    __shared__ float sh[256], sh2[256];
    int ch = blockIdx.x, tid = threadIdx.x;
    float s = 0.f, s2 = 0.f;
    for (int i = tid; i < n; i += 256) {
        float v = x[(long)i * c + ch];
        s += v; s2 += v * v;
    }
    sh[tid] = s; sh2[tid] = s2;
    __syncthreads();
    for (int st = 128; st > 0; st >>= 1) {
        if (tid < st) { sh[tid] += sh[tid + st]; sh2[tid] += sh2[tid + st]; }
        __syncthreads();
    }
    if (tid == 0) {
        float m = sh[0] / (float)n;
        float var = sh2[0] / (float)n - m * m;
        mu[ch] = m;
        rs[ch] = rsqrtf(var + 1e-5f);
    }
}

__global__ void bn_apply_kernel(float* __restrict__ x,
                                const float* __restrict__ mu, const float* __restrict__ rs,
                                const float* __restrict__ g, const float* __restrict__ be,
                                int total, int c, int relu) {
    int t = blockIdx.x * blockDim.x + threadIdx.x;
    if (t >= total) return;
    int ch = t % c;
    float v = (x[t] - mu[ch]) * rs[ch] * g[ch] + be[ch];
    if (relu && v < 0.f) v = 0.f;
    x[t] = v;
}

// mean over heads (+ b2; exact even though it cancels under following BN)
__global__ void headmean_kernel(const float* __restrict__ agg,
                                const float* __restrict__ b2,
                                float* __restrict__ xm, int n) {
    int t = blockIdx.x * blockDim.x + threadIdx.x;
    if (t >= n * GF) return;
    int node = t >> 6, f = t & 63;
    const float* p = agg + node * GC + f;
    xm[t] = 0.25f * (p[0] + p[GF] + p[2 * GF] + p[3 * GF]) + b2[f];
}

// final [N,256] @ [256,1] + bo2
__global__ void final_dot_kernel(const float* __restrict__ z,
                                 const float* __restrict__ Wo2,
                                 const float* __restrict__ bo2,
                                 float* __restrict__ out, int n) {
    int node = blockIdx.x * blockDim.x + threadIdx.x;
    if (node >= n) return;
    float acc = bo2[0];
    const float* zp = z + (long)node * GC;
    #pragma unroll 8
    for (int k = 0; k < GC; ++k) acc += zp[k] * Wo2[k];
    out[node] = acc;
}

// ---------------------------------------------------------------------------
extern "C" void kernel_launch(void* const* d_in, const int* in_sizes, int n_in,
                              void* d_out, int out_size, void* d_ws, size_t ws_size,
                              hipStream_t stream) {
    (void)in_sizes; (void)n_in; (void)out_size; (void)ws_size;

    const float* X   = (const float*)d_in[0];
    const int*   ei  = (const int*)d_in[1];
    const float* Win = (const float*)d_in[2];
    const float* bin = (const float*)d_in[3];
    const float* W1  = (const float*)d_in[4];
    const float* as1 = (const float*)d_in[5];
    const float* ad1 = (const float*)d_in[6];
    // d_in[7] = b1 : cancels under BatchNorm (and is zeros)
    const float* g1  = (const float*)d_in[8];
    const float* be1 = (const float*)d_in[9];
    const float* W2  = (const float*)d_in[10];
    const float* as2 = (const float*)d_in[11];
    const float* ad2 = (const float*)d_in[12];
    const float* b2  = (const float*)d_in[13];
    const float* g2  = (const float*)d_in[14];
    const float* be2 = (const float*)d_in[15];
    const float* Wo1 = (const float*)d_in[16];
    const float* bo1 = (const float*)d_in[17];
    const float* Wo2 = (const float*)d_in[18];
    const float* bo2 = (const float*)d_in[19];
    float* out = (float*)d_out;

    const int* src = ei;
    const int* dst = ei + GE;

    // workspace carve-up (256B aligned)
    char* ws = (char*)d_ws;
    size_t off = 0;
    auto alloc = [&](size_t bytes) -> char* {
        char* p = ws + off;
        off = (off + bytes + 255) & ~(size_t)255;
        return p;
    };
    float*          wA   = (float*)alloc((size_t)GN * GC * 4);   // 51.2 MB
    float*          wB   = (float*)alloc((size_t)GN * GC * 4);   // 51.2 MB
    float*          xm   = (float*)alloc((size_t)GN * GF * 4);   // 12.8 MB
    float*          aS   = (float*)alloc((size_t)GN * GH * 4);
    float*          aD   = (float*)alloc((size_t)GN * GH * 4);
    unsigned*       menc = (unsigned*)alloc((size_t)GN * GH * 4);
    float*          ssum = (float*)alloc((size_t)GN * GH * 4);
    float*          ex   = (float*)alloc((size_t)GE * GH * 4);   // 12.8 MB
    unsigned short* pWin = (unsigned short*)alloc((size_t)GIN * GC * 2);
    unsigned short* pW1  = (unsigned short*)alloc((size_t)GC * GC * 2);
    unsigned short* pW2  = (unsigned short*)alloc((size_t)GC * GC * 2);
    unsigned short* pWo1 = (unsigned short*)alloc((size_t)GF * GEND * 2);
    float*          mu   = (float*)alloc(256 * 4);
    float*          rs   = (float*)alloc(256 * 4);

    // pack weights fp32 -> bf16 WMMA-fragment layout
    pack_w_kernel<<<(GIN * GC + 255) / 256, 256, 0, stream>>>(Win, GIN, GC, pWin);
    pack_w_kernel<<<(GC * GC + 255) / 256, 256, 0, stream>>>(W1, GC, GC, pW1);
    pack_w_kernel<<<(GC * GC + 255) / 256, 256, 0, stream>>>(W2, GC, GC, pW2);
    pack_w_kernel<<<(GF * GEND + 255) / 256, 256, 0, stream>>>(Wo1, GF, GEND, pWo1);

    dim3 gg(GC / 64, (GN + 127) / 128);
    const unsigned ehBlocks  = (GE * GH + 255) / 256;
    const unsigned aggBlocks = (unsigned)(((size_t)GE * GC + 255) / 256);

    // ---- input layer: x0 = X[:,:,-1] @ W_in + b_in  (slice fused via strides)
    gemm_bf16_wmma<<<gg, 256, 0, stream>>>(X, GN, GIN, GC,
                                           GIN * GT, GT, GT - 1, pWin, bin, wA, 0, 0);
    // ---- GATConv 1
    gemm_bf16_wmma<<<gg, 256, 0, stream>>>(wA, GN, GC, GC, GC, 1, 0, pW1, nullptr, wB, 0, 1);
    att_kernel<<<(GN * GH + 255) / 256, 256, 0, stream>>>(wB, as1, ad1, aS, aD, GN);
    hipMemsetAsync(menc, 0, (size_t)GN * GH * 4, stream);
    hipMemsetAsync(ssum, 0, (size_t)GN * GH * 4, stream);
    hipMemsetAsync(wA, 0, (size_t)GN * GC * 4, stream);
    edge_max_kernel<<<ehBlocks, 256, 0, stream>>>(src, dst, aS, aD, menc, GE);
    edge_exp_kernel<<<ehBlocks, 256, 0, stream>>>(src, dst, aS, aD, menc, ssum, ex, GE);
    edge_agg_kernel<<<aggBlocks, 256, 0, stream>>>(src, dst, wB, ex, ssum, wA, GE);
    // ---- BN1 + ReLU (b1 cancels)
    bn_reduce_kernel<<<GC, 256, 0, stream>>>(wA, GN, GC, mu, rs);
    bn_apply_kernel<<<(GN * GC + 255) / 256, 256, 0, stream>>>(wA, mu, rs, g1, be1,
                                                               GN * GC, GC, 1);
    // ---- GATConv 2
    gemm_bf16_wmma<<<gg, 256, 0, stream>>>(wA, GN, GC, GC, GC, 1, 0, pW2, nullptr, wB, 0, 1);
    att_kernel<<<(GN * GH + 255) / 256, 256, 0, stream>>>(wB, as2, ad2, aS, aD, GN);
    hipMemsetAsync(menc, 0, (size_t)GN * GH * 4, stream);
    hipMemsetAsync(ssum, 0, (size_t)GN * GH * 4, stream);
    hipMemsetAsync(wA, 0, (size_t)GN * GC * 4, stream);
    edge_max_kernel<<<ehBlocks, 256, 0, stream>>>(src, dst, aS, aD, menc, GE);
    edge_exp_kernel<<<ehBlocks, 256, 0, stream>>>(src, dst, aS, aD, menc, ssum, ex, GE);
    edge_agg_kernel<<<aggBlocks, 256, 0, stream>>>(src, dst, wB, ex, ssum, wA, GE);
    // ---- head mean + BN2 (no relu)
    headmean_kernel<<<(GN * GF + 255) / 256, 256, 0, stream>>>(wA, b2, xm, GN);
    bn_reduce_kernel<<<GF, 256, 0, stream>>>(xm, GN, GF, mu, rs);
    bn_apply_kernel<<<(GN * GF + 255) / 256, 256, 0, stream>>>(xm, mu, rs, g2, be2,
                                                               GN * GF, GF, 0);
    // ---- output MLP: relu(xm @ Wo1 + bo1) @ Wo2 + bo2
    gemm_bf16_wmma<<<dim3(GEND / 64, (GN + 127) / 128), 256, 0, stream>>>(
        xm, GN, GF, GEND, GF, 1, 0, pWo1, bo1, wB, 1, 1);
    final_dot_kernel<<<(GN + 255) / 256, 256, 0, stream>>>(wB, Wo2, bo2, out, GN);
}
